// reversi_model_5265629905180
// MI455X (gfx1250) — matrix operand order repaired and checked by reading, not compiled, not run on hardware
//
#include <hip/hip_runtime.h>
#include <math.h>

// ---------------------------------------------------------------------------
// Model dims
// ---------------------------------------------------------------------------
#define Hdim 128          // hidden
#define Sdim 65           // sequence (64 inputs + bias)
#define SP   80           // S padded to multiple of 16 (M/N dim)
#define TP   96           // S padded to multiple of 32 (K dim for sm @ v^T)

typedef __bf16 bf16;
typedef __bf16 v16bf __attribute__((ext_vector_type(16)));
typedef float  v8f   __attribute__((ext_vector_type(8)));

// ---------------------------------------------------------------------------
// WMMA operand loaders.  All matrices are stored so the K dimension is
// contiguous:  A row-major (M x K, stride lda), B as B^T row-major (N x K,
// stride ldbt).  Layouts follow CDNA5 ISA §7.12.2 (wave32).
// ---------------------------------------------------------------------------
__device__ inline v16bf load_a(const bf16* A, int lda, int m0, int k0, int lane) {
    int m  = m0 + (lane & 15);
    int h2 = (lane >> 4) << 2;                 // unsigned-index offset: 0 or 4
    const unsigned* p = (const unsigned*)(A + m * lda + k0);
    union { v16bf v; unsigned u[8]; } r;
#pragma unroll
    for (int j = 0; j < 4; ++j) {
        r.u[j]     = p[h2 + j];                // K = 8*half + 2j   (+1)
        r.u[4 + j] = p[8 + h2 + j];            // K = 16 + 8*half + 2j (+1)
    }
    return r.v;
}

__device__ inline v16bf load_b(const bf16* Bt, int ldbt, int k0, int n0, int lane) {
    int n    = n0 + (lane & 15);
    int koff = (lane >> 4) << 3;               // 0 or 8
    const unsigned* p = (const unsigned*)(Bt + n * ldbt + k0 + koff);
    union { v16bf v; unsigned u[8]; } r;
#pragma unroll
    for (int j = 0; j < 4; ++j) { r.u[j] = p[j]; r.u[4 + j] = p[8 + j]; } // K = koff + 2j (+1), 16 + koff + 2j (+1)
    return r.v;
}

// C/D 16x16 f32 layout: lane -> N = lane&15 ; VGPR r -> M = 8*(lane>>4) + r
__device__ inline void store_f32(float* C, int ldc, int m0, int n0, v8f a, int lane) {
    int n  = n0 + (lane & 15);
    int mb = m0 + ((lane >> 4) << 3);
#pragma unroll
    for (int r = 0; r < 8; ++r) C[(mb + r) * ldc + n] = a[r];
}
__device__ inline void store_bf16(bf16* C, int ldc, int m0, int n0, v8f a, int lane) {
    int n  = n0 + (lane & 15);
    int mb = m0 + ((lane >> 4) << 3);
#pragma unroll
    for (int r = 0; r < 8; ++r) C[(mb + r) * ldc + n] = (bf16)a[r];
}
__device__ inline void store_bf16_T(bf16* Ct, int ldct, int m0, int n0, v8f a, int lane) {
    int n  = n0 + (lane & 15);
    int mb = m0 + ((lane >> 4) << 3);
#pragma unroll
    for (int r = 0; r < 8; ++r) Ct[n * ldct + (mb + r)] = (bf16)a[r];
}
__device__ inline void store_tanh_bf16(bf16* C, int ldc, int m0, int n0, v8f a, int lane) {
    int n  = n0 + (lane & 15);
    int mb = m0 + ((lane >> 4) << 3);
#pragma unroll
    for (int r = 0; r < 8; ++r) C[(mb + r) * ldc + n] = (bf16)tanhf(a[r]);
}

// Accumulate one 16x16 output tile over K (multiple of 32) with bf16 WMMA.
__device__ inline v8f mm_tile(const bf16* A, int lda, const bf16* Bt, int ldbt,
                              int m0, int n0, int K, int lane) {
    v8f acc = {0.f, 0.f, 0.f, 0.f, 0.f, 0.f, 0.f, 0.f};
    for (int k0 = 0; k0 < K; k0 += 32) {
        v16bf a = load_a(A, lda, m0, k0, lane);
        v16bf b = load_b(Bt, ldbt, k0, n0, lane);
        acc = __builtin_amdgcn_wmma_f32_16x16x32_bf16(
            false, a, false, b, (short)0, acc, false, false);
    }
    return acc;
}

// ---------------------------------------------------------------------------
// Weight pre-pass: f32 -> bf16 (plain and transposed) into workspace.
// ---------------------------------------------------------------------------
__global__ void cvt_plain(const float* __restrict__ src, bf16* __restrict__ dst, int n) {
    int i = blockIdx.x * 256 + threadIdx.x;
    if (i < n) dst[i] = (bf16)src[i];
}
// src is (C x O) row-major, dst is (O x C) row-major (i.e. src^T)
__global__ void cvt_trans(const float* __restrict__ src, bf16* __restrict__ dst,
                          int C, int O) {
    int i = blockIdx.x * 256 + threadIdx.x;
    if (i < C * O) {
        int o = i / C, c = i % C;
        dst[i] = (bf16)src[c * O + o];
    }
}

// ---------------------------------------------------------------------------
// Fused model: one block (256 thr = 8 wave32) per batch element.
//
// LDS carve (bytes):
//   inT   [SP x H]  bf16  20480   B-operand of projections (layer input, transposed)
//   qT    [SP x H]  bf16  20480   q^T  (rows t, K=i contiguous)
//   kT    [SP x H]  bf16  20480   k^T  (rows s) ; reused as sm (80x96 bf16) after kq
//   vb    [H  x TP] bf16  24576   v    (rows i, K=t contiguous)
//   kqf   [SP x SP] f32   25600   attention logits
//   catbf [SP x 2H] bf16  40960   concat(att_h0, att_h1), K=c contiguous
//   xs    [SP]      f32     320
//   red   [64]      f32     256
// ---------------------------------------------------------------------------
#define OFF_INT   0
#define OFF_QT    20480
#define OFF_KT    40960
#define OFF_VB    61440
#define OFF_KQF   86016
#define OFF_CAT   111616
#define OFF_XS    152576
#define OFF_RED   152896
#define SMEM_SZ   153152

#define WSTRIDE   (6 * Hdim * Hdim + 2 * Hdim * Hdim)   // 6 proj mats + l^T per layer

__global__ __launch_bounds__(256) void fused_model(
    const float* __restrict__ inputs,   // (B, 64)
    const float* __restrict__ emb,      // (H, S) row-major
    const bf16*  __restrict__ wsW,      // bf16 weights (see kernel_launch)
    const float* __restrict__ l3,       // (128, 64)
    const float* __restrict__ l4,       // (64,)
    float*       __restrict__ out,      // (B,)
    int B)
{
    __shared__ __align__(16) char smem[SMEM_SZ];
    bf16*  inT   = (bf16*)(smem + OFF_INT);
    bf16*  qT    = (bf16*)(smem + OFF_QT);
    bf16*  kT    = (bf16*)(smem + OFF_KT);
    bf16*  smbf  = (bf16*)(smem + OFF_KT);     // aliases kT (dead after kq GEMM)
    bf16*  vb    = (bf16*)(smem + OFF_VB);
    float* kqf   = (float*)(smem + OFF_KQF);
    bf16*  catbf = (bf16*)(smem + OFF_CAT);
    float* xs    = (float*)(smem + OFF_XS);
    float* red   = (float*)(smem + OFF_RED);

    const int b    = blockIdx.x;
    const int tid  = threadIdx.x;
    const int lane = tid & 31;
    const int wave = tid >> 5;
    if (b >= B) return;

    // x = [inputs_b, 1], padded with zeros to SP
    if (tid < SP) xs[tid] = (tid < 64) ? inputs[b * 64 + tid]
                                       : (tid == 64 ? 1.0f : 0.0f);
    __syncthreads();

    // Layer-1 input, already transposed: inT[s][j] = emb[j][s] * x[s]
    for (int idx = tid; idx < SP * Hdim; idx += 256) {
        int s = idx >> 7, j = idx & 127;
        float v = (s < Sdim) ? emb[j * Sdim + s] * xs[s] : 0.0f;
        inT[idx] = (bf16)v;
    }
    __syncthreads();

    for (int layer = 0; layer < 2; ++layer) {
        const bf16* Wbase = wsW + layer * WSTRIDE;
        const bf16* lT    = Wbase + 6 * Hdim * Hdim;   // (128 x 256) = linear^T

        for (int head = 0; head < 2; ++head) {
            const bf16* Wk = Wbase + (head * 3 + 0) * Hdim * Hdim;
            const bf16* Wq = Wbase + (head * 3 + 1) * Hdim * Hdim;
            const bf16* Wv = Wbase + (head * 3 + 2) * Hdim * Hdim;

            // zero v K-padding columns [80, 96)
            for (int idx = tid; idx < Hdim * 16; idx += 256) {
                int i = idx >> 4, t = SP + (idx & 15);
                vb[i * TP + t] = (bf16)0.0f;
            }
            __syncthreads();

            // Projections: C[i][s] = sum_j W[i][j] * inT_B[j][s]
            // (A = W row-major, B^T = inT). 8x5 = 40 tiles.
            for (int t = wave; t < 40; t += 8) {
                int mt = (t / 5) * 16, nt = (t % 5) * 16;
                v8f ak = mm_tile(Wk, Hdim, inT, Hdim, mt, nt, Hdim, lane);
                v8f aq = mm_tile(Wq, Hdim, inT, Hdim, mt, nt, Hdim, lane);
                v8f av = mm_tile(Wv, Hdim, inT, Hdim, mt, nt, Hdim, lane);
                store_bf16_T(kT, Hdim, mt, nt, ak, lane);  // kT[s][i]
                store_bf16_T(qT, Hdim, mt, nt, aq, lane);  // qT[t][i]
                store_bf16  (vb, TP,   mt, nt, av, lane);  // v [i][t]
            }
            __syncthreads();

            // Scores: kq[s][t] = sum_i kT[s][i] * qT[t][i].  5x5 = 25 tiles.
            for (int t = wave; t < 25; t += 8) {
                int mt = (t / 5) * 16, nt = (t % 5) * 16;
                v8f acc = mm_tile(kT, Hdim, qT, Hdim, mt, nt, Hdim, lane);
                store_f32(kqf, SP, mt, nt, acc, lane);
            }
            __syncthreads();

            // Row softmax over t < 65; zero padding (kT is dead -> smbf).
            if (tid < SP) {
                int s = tid;
                if (s < Sdim) {
                    float mx = -1e30f;
                    for (int t = 0; t < Sdim; ++t) mx = fmaxf(mx, kqf[s * SP + t]);
                    float sum = 0.0f;
                    for (int t = 0; t < Sdim; ++t) sum += __expf(kqf[s * SP + t] - mx);
                    float inv = 1.0f / sum;
                    for (int t = 0; t < Sdim; ++t)
                        smbf[s * TP + t] = (bf16)(__expf(kqf[s * SP + t] - mx) * inv);
                    for (int t = Sdim; t < TP; ++t) smbf[s * TP + t] = (bf16)0.0f;
                } else {
                    for (int t = 0; t < TP; ++t) smbf[s * TP + t] = (bf16)0.0f;
                }
            }
            __syncthreads();

            // Apply: att[s][i] = sum_t sm[s][t] * v[i][t].  5x8 = 40 tiles.
            // Store into concat buffer (K=c contiguous for the linear GEMM).
            for (int t = wave; t < 40; t += 8) {
                int mt = (t / 8) * 16, nt = (t % 8) * 16;
                v8f acc = mm_tile(smbf, TP, vb, TP, mt, nt, TP, lane);
                store_bf16(catbf + head * Hdim, 2 * Hdim, mt, nt, acc, lane);
            }
            __syncthreads();
        } // head

        // z[s][o] = tanh( sum_c cat[s][c] * linear[c][o] ), K = 256.
        // Output becomes next layer's transposed input (swapaxes is free).
        for (int t = wave; t < 40; t += 8) {
            int mt = (t / 8) * 16, nt = (t % 8) * 16;
            v8f acc = mm_tile(catbf, 2 * Hdim, lT, 2 * Hdim, mt, nt, 2 * Hdim, lane);
            store_tanh_bf16(inT, Hdim, mt, nt, acc, lane);
        }
        __syncthreads();
    } // layer

    // Final head on row s = 64:  tanh(h @ l3) @ l4
    if (tid < 64) {
        float acc = 0.0f;
        for (int c = 0; c < Hdim; ++c)
            acc += (float)inT[64 * Hdim + c] * l3[c * 64 + tid];
        red[tid] = tanhf(acc) * l4[tid];
    }
    __syncthreads();
    if (tid == 0) {
        float s = 0.0f;
        for (int o = 0; o < 64; ++o) s += red[o];
        out[b] = s;
    }
}

// ---------------------------------------------------------------------------
// Launch
// ---------------------------------------------------------------------------
extern "C" void kernel_launch(void* const* d_in, const int* in_sizes, int n_in,
                              void* d_out, int out_size, void* d_ws, size_t ws_size,
                              hipStream_t stream) {
    (void)n_in; (void)out_size; (void)ws_size;
    const int HH = Hdim * Hdim;          // 16384
    const int B  = in_sizes[0] / 64;     // 8192

    bf16* ws = (bf16*)d_ws;
    // d_in order: 0 inputs, 1 emb, 2-4 wk/wq/wv11, 5-7 wk/wq/wv12, 8 linear1,
    //             9-11 wk/wq/wv21, 12-14 wk/wq/wv22, 15 linear2, 16 l3, 17 l4
    const int wIdx[12] = {2, 3, 4, 5, 6, 7, 9, 10, 11, 12, 13, 14};

    for (int layer = 0; layer < 2; ++layer) {
        bf16* base = ws + layer * WSTRIDE;
        for (int j = 0; j < 6; ++j)
            cvt_plain<<<(HH + 255) / 256, 256, 0, stream>>>(
                (const float*)d_in[wIdx[layer * 6 + j]], base + j * HH, HH);
        cvt_trans<<<(2 * HH + 255) / 256, 256, 0, stream>>>(
            (const float*)d_in[layer == 0 ? 8 : 15], base + 6 * HH, 256, 128);
    }

    fused_model<<<B, 256, 0, stream>>>(
        (const float*)d_in[0], (const float*)d_in[1], ws,
        (const float*)d_in[16], (const float*)d_in[17],
        (float*)d_out, B);
}
